// ResidualGatedFusion_41979010351484
// MI455X (gfx1250) — matrix operand order repaired and checked by reading, not compiled, not run on hardware
//
#include <hip/hip_runtime.h>
#include <math.h>

// ---------------------------------------------------------------------------
// ResidualGatedFusion forward for MI455X (gfx1250, wave32, WMMA).
// GEMMs: bf16 operands (pre-converted once; weights pre-transposed to [N,K]),
// v_wmma_f32_16x16x32_bf16 accumulate in f32, LDS double-buffered with
// GLOBAL_LOAD_ASYNC_TO_LDS_B128. Elementwise/LN/BN stay f32.
// ---------------------------------------------------------------------------

#define BDIM 8192
#define DDIM 2048
#define HDIM 512
#define LN_EPS 1e-5f

typedef __attribute__((ext_vector_type(16))) __bf16 v16bf;
typedef __attribute__((ext_vector_type(8)))  float  v8f;
typedef __attribute__((ext_vector_type(4)))  unsigned int u32x4;

// async-copy pointer types per clang builtin signature:
//   (v4i addrspace(1)*, v4i addrspace(3)*, imm offset, imm cpol)
typedef int v4i_t __attribute__((vector_size(16)));
typedef __attribute__((address_space(1))) v4i_t* as1_v4i;
typedef __attribute__((address_space(3))) v4i_t* as3_v4i;

#if defined(__has_builtin)
#if __has_builtin(__builtin_amdgcn_global_load_async_to_lds_b128)
#define HAS_ASYNC_LDS 1
#endif
#endif
#ifndef HAS_ASYNC_LDS
#define HAS_ASYNC_LDS 0
#endif

#if defined(__has_builtin)
#if __has_builtin(__builtin_amdgcn_s_wait_asynccnt)
#define WAIT_ASYNC(n) __builtin_amdgcn_s_wait_asynccnt(n)
#endif
#endif
#ifndef WAIT_ASYNC
#define WAIT_ASYNC(n) asm volatile("s_wait_asynccnt %0" ::"n"(n) : "memory")
#endif

__device__ __forceinline__ unsigned short f2bf(float a) {
  unsigned int u = __float_as_uint(a);
  u += 0x7FFFu + ((u >> 16) & 1u);  // round-to-nearest-even
  return (unsigned short)(u >> 16);
}

__device__ __forceinline__ float gelu_exact(float x) {
  return 0.5f * x * (1.0f + erff(x * 0.7071067811865475f));
}

// ---------------------------------------------------------------------------
// Tiled bf16 WMMA GEMM: C[M,N] = A[M,K] @ Bt[N,K]^T + bias[N]
// A, Bt are bf16; C is f32 (OUTBF=0) or bf16 (OUTBF=1). EPI=1 fuses GELU.
// 256 threads (8 wave32) per 128x128 C tile; waves 4(M) x 2(N); K-step 32.
// ---------------------------------------------------------------------------
#define TLDA 40  // padded LDS row stride (elements); 80B rows keep 16B align

template <int EPI, int OUTBF>
__global__ __launch_bounds__(256) void gemm_bf16_kernel(
    const unsigned short* __restrict__ A, const unsigned short* __restrict__ Bt,
    const float* __restrict__ bias, void* __restrict__ Cout,
    int M, int N, int K) {
  constexpr int NB = HAS_ASYNC_LDS ? 2 : 1;
  __shared__ __attribute__((aligned(16))) unsigned short Asm[NB][128 * TLDA];
  __shared__ __attribute__((aligned(16))) unsigned short Bsm[NB][128 * TLDA];

  const int tid  = threadIdx.x;
  const int lane = tid & 31;
  const int w    = tid >> 5;
  const int wm   = w >> 1;      // 0..3
  const int wn   = w & 1;       // 0..1
  const int lm   = lane & 15;
  const int kg   = lane >> 4;   // half-wave selector per ISA 7.12.2 layouts
  const int m0   = blockIdx.y * 128;
  const int n0   = blockIdx.x * 128;

  const v8f vzero = {0.f, 0.f, 0.f, 0.f, 0.f, 0.f, 0.f, 0.f};
  v8f acc[2][4];
#pragma unroll
  for (int i = 0; i < 2; ++i)
#pragma unroll
    for (int j = 0; j < 4; ++j) acc[i][j] = vzero;

  const int nk = K >> 5;

  // per-thread staging coordinates: 2 chunks of 8 bf16 for A, same for B
  // chunk idx in [0,512): row = idx>>2, col8 = (idx&3)*8  (128 rows x 32 k)
  auto compute = [&](const unsigned short* As, const unsigned short* Bs) {
    union Frag { u32x4 u[2]; v16bf v; };
    Frag a[2], b[4];
#pragma unroll
    for (int i = 0; i < 2; ++i) {
      // A 16x32: lane holds K [8*kg,8*kg+8) and [16+8*kg,16+8*kg+8)
      const unsigned short* p = &As[(wm * 32 + i * 16 + lm) * TLDA + (kg << 3)];
      a[i].u[0] = *(const u32x4*)(p);
      a[i].u[1] = *(const u32x4*)(p + 16);
    }
#pragma unroll
    for (int j = 0; j < 4; ++j) {
      // B 32x16: lane holds K [8*kg,8*kg+8) and [16+8*kg,16+8*kg+8) of one column
      const unsigned short* p = &Bs[(wn * 64 + j * 16 + lm) * TLDA + (kg << 3)];
      b[j].u[0] = *(const u32x4*)(p);
      b[j].u[1] = *(const u32x4*)(p + 16);
    }
#pragma unroll
    for (int i = 0; i < 2; ++i)
#pragma unroll
      for (int j = 0; j < 4; ++j)
        acc[i][j] = __builtin_amdgcn_wmma_f32_16x16x32_bf16(
            false, a[i].v, false, b[j].v, (short)0, acc[i][j], false, false);
  };

#if HAS_ASYNC_LDS
  auto stage_async = [&](int kt, int bb) {
#pragma unroll
    for (int i = 0; i < 2; ++i) {
      int idx = tid + i * 256;
      int rr = idx >> 2;
      int cc = (idx & 3) << 3;
      __builtin_amdgcn_global_load_async_to_lds_b128(
          (as1_v4i)(A + (size_t)(m0 + rr) * K + (kt << 5) + cc),
          (as3_v4i)(&Asm[bb][rr * TLDA + cc]), 0, 0);
      __builtin_amdgcn_global_load_async_to_lds_b128(
          (as1_v4i)(Bt + (size_t)(n0 + rr) * K + (kt << 5) + cc),
          (as3_v4i)(&Bsm[bb][rr * TLDA + cc]), 0, 0);
    }
  };
  stage_async(0, 0);
  for (int kt = 0; kt < nk; ++kt) {
    const int buf = kt & 1;
    if (kt + 1 < nk) {
      stage_async(kt + 1, buf ^ 1);  // prefetch next tile into other buffer
      WAIT_ASYNC(4);                 // this wave's 4 copies for tile kt done
    } else {
      WAIT_ASYNC(0);
    }
    __syncthreads();
    compute(Asm[buf], Bsm[buf]);
    __syncthreads();  // all waves done reading before buffer is rewritten
  }
#else
  for (int kt = 0; kt < nk; ++kt) {
    u32x4 ra[2], rb[2];
#pragma unroll
    for (int i = 0; i < 2; ++i) {  // issue all global loads first
      int idx = tid + i * 256;
      int rr = idx >> 2;
      int cc = (idx & 3) << 3;
      ra[i] = *(const u32x4*)(A + (size_t)(m0 + rr) * K + (kt << 5) + cc);
      rb[i] = *(const u32x4*)(Bt + (size_t)(n0 + rr) * K + (kt << 5) + cc);
    }
#pragma unroll
    for (int i = 0; i < 2; ++i) {
      int idx = tid + i * 256;
      int rr = idx >> 2;
      int cc = (idx & 3) << 3;
      *(u32x4*)(&Asm[0][rr * TLDA + cc]) = ra[i];
      *(u32x4*)(&Bsm[0][rr * TLDA + cc]) = rb[i];
    }
    __syncthreads();
    compute(Asm[0], Bsm[0]);
    __syncthreads();
  }
#endif

  // epilogue: C layout = VGPR r -> M = 16*i + 8*kg + r; lanes carry N
#pragma unroll
  for (int i = 0; i < 2; ++i) {
#pragma unroll
    for (int j = 0; j < 4; ++j) {
      const int n  = n0 + wn * 64 + j * 16 + lm;
      const float bv = bias[n];
#pragma unroll
      for (int r = 0; r < 8; ++r) {
        const int m = m0 + wm * 32 + i * 16 + (kg << 3) + r;
        float v = acc[i][j][r] + bv;
        if (EPI == 1) v = gelu_exact(v);
        if (OUTBF)
          ((unsigned short*)Cout)[(size_t)m * N + n] = f2bf(v);
        else
          ((float*)Cout)[(size_t)m * N + n] = v;
      }
    }
  }
}

// ---------------------------------------------------------------------------
// Pre-conversion kernels
// ---------------------------------------------------------------------------

// W[K,N] f32 -> Wt[N,K] bf16, 32x32 LDS-tiled transpose
__global__ __launch_bounds__(256) void wt_transpose_kernel(
    const float* __restrict__ W, unsigned short* __restrict__ Wt, int K, int N) {
  __shared__ float tile[32][33];
  const int kb = blockIdx.y * 32, nb = blockIdx.x * 32;
  const int tx = threadIdx.x & 31, ty = threadIdx.x >> 5;  // ty: 0..7
  for (int i = ty; i < 32; i += 8)
    tile[i][tx] = W[(size_t)(kb + i) * N + nb + tx];
  __syncthreads();
  for (int i = ty; i < 32; i += 8)
    Wt[(size_t)(nb + i) * K + kb + tx] = f2bf(tile[tx][i]);
}

__global__ void cvt_bf16_kernel(const float* __restrict__ in,
                                unsigned short* __restrict__ out, size_t n) {
  size_t i = (size_t)blockIdx.x * 256 + threadIdx.x;
  if (i < n) out[i] = f2bf(in[i]);
}

__global__ void concat_bf16_kernel(const float* __restrict__ f1,
                                   const float* __restrict__ f2,
                                   unsigned short* __restrict__ out) {
  size_t idx = (size_t)blockIdx.x * 256 + threadIdx.x;
  size_t total = (size_t)BDIM * 2 * DDIM;
  if (idx >= total) return;
  size_t row = idx / (2 * DDIM);
  size_t col = idx - row * (2 * DDIM);
  float v = (col < DDIM) ? f1[row * DDIM + col] : f2[row * DDIM + col - DDIM];
  out[idx] = f2bf(v);
}

// ---------------------------------------------------------------------------
// LayerNorm: out{f32,bf16} = [gelu](LN(x (+res))); block per row; in-place ok
// ---------------------------------------------------------------------------
__global__ __launch_bounds__(256) void ln_kernel(
    const float* __restrict__ x, const float* __restrict__ res,
    const float* __restrict__ g, const float* __restrict__ b,
    float* __restrict__ outf, unsigned short* __restrict__ outb,
    int n, int do_gelu) {
  const int row = blockIdx.x;
  const float* xr = x + (size_t)row * n;
  const float* rr = res ? res + (size_t)row * n : nullptr;
  __shared__ float s1[256], s2[256];
  float a0 = 0.f, a1 = 0.f;
  for (int c = threadIdx.x; c < n; c += 256) {
    float v = xr[c] + (rr ? rr[c] : 0.0f);
    a0 += v;
    a1 += v * v;
  }
  s1[threadIdx.x] = a0;
  s2[threadIdx.x] = a1;
  __syncthreads();
  for (int s = 128; s > 0; s >>= 1) {
    if (threadIdx.x < s) {
      s1[threadIdx.x] += s1[threadIdx.x + s];
      s2[threadIdx.x] += s2[threadIdx.x + s];
    }
    __syncthreads();
  }
  const float mean = s1[0] / (float)n;
  const float var  = s2[0] / (float)n - mean * mean;
  const float inv  = rsqrtf(var + LN_EPS);
  for (int c = threadIdx.x; c < n; c += 256) {
    float v = xr[c] + (rr ? rr[c] : 0.0f);
    float y = (v - mean) * inv * g[c] + b[c];
    if (do_gelu) y = gelu_exact(y);
    if (outf) outf[(size_t)row * n + c] = y;
    if (outb) outb[(size_t)row * n + c] = f2bf(y);
  }
}

__global__ void gate_combine_kernel(const float* __restrict__ gl,
                                    const float* __restrict__ f1,
                                    const float* __restrict__ f2,
                                    float* __restrict__ gf,
                                    unsigned short* __restrict__ gfb) {
  size_t idx = (size_t)blockIdx.x * 256 + threadIdx.x;
  if (idx >= (size_t)BDIM * DDIM) return;
  float s = 1.0f / (1.0f + expf(-gl[idx]));
  float v = s * f1[idx] + (1.0f - s) * f2[idx];
  gf[idx] = v;
  gfb[idx] = f2bf(v);
}

// 2-token attention: out = gf + softmax([q.k1,q.k2]/sqrt(D)) . [v1,v2]
__global__ __launch_bounds__(256) void attn_kernel(
    const float* __restrict__ q, const float* __restrict__ k1,
    const float* __restrict__ k2, const float* __restrict__ v1,
    const float* __restrict__ v2, const float* __restrict__ gf,
    float* __restrict__ out) {
  const int row = blockIdx.x;
  const size_t base = (size_t)row * DDIM;
  __shared__ float s1[256], s2[256];
  float a1 = 0.f, a2 = 0.f;
  for (int c = threadIdx.x; c < DDIM; c += 256) {
    float qv = q[base + c];
    a1 += qv * k1[base + c];
    a2 += qv * k2[base + c];
  }
  s1[threadIdx.x] = a1;
  s2[threadIdx.x] = a2;
  __syncthreads();
  for (int s = 128; s > 0; s >>= 1) {
    if (threadIdx.x < s) {
      s1[threadIdx.x] += s1[threadIdx.x + s];
      s2[threadIdx.x] += s2[threadIdx.x + s];
    }
    __syncthreads();
  }
  const float scale = 0.022097086912079608f;  // 1/sqrt(2048)
  float e1 = s1[0] * scale, e2 = s2[0] * scale;
  float mx = fmaxf(e1, e2);
  float x1 = expf(e1 - mx), x2 = expf(e2 - mx);
  float w1 = x1 / (x1 + x2), w2 = 1.0f - w1;
  for (int c = threadIdx.x; c < DDIM; c += 256)
    out[base + c] = gf[base + c] + w1 * v1[base + c] + w2 * v2[base + c];
}

__global__ void zero_kernel(float* __restrict__ p, int n) {
  int i = blockIdx.x * 256 + threadIdx.x;
  if (i < n) p[i] = 0.f;
}

__global__ __launch_bounds__(256) void bn_count_kernel(
    const int* __restrict__ ids, float* __restrict__ cnt) {
  __shared__ int c[3];
  if (threadIdx.x < 3) c[threadIdx.x] = 0;
  __syncthreads();
  for (int r = threadIdx.x; r < BDIM; r += 256) atomicAdd(&c[ids[r]], 1);
  __syncthreads();
  if (threadIdx.x < 3) cnt[threadIdx.x] = (float)c[threadIdx.x];
}

// Per-aspect column sums/sumsq; grid = (D/256, B/256); f32 global atomics.
__global__ __launch_bounds__(256) void bn_stats_kernel(
    const float* __restrict__ x, const int* __restrict__ ids,
    float* __restrict__ sum, float* __restrict__ sq) {
  const int col = blockIdx.x * 256 + threadIdx.x;
  const int r0  = blockIdx.y * 256;
  float s0 = 0.f, s1 = 0.f, s2 = 0.f, q0 = 0.f, q1 = 0.f, q2 = 0.f;
  for (int r = r0; r < r0 + 256; ++r) {
    float v = x[(size_t)r * DDIM + col];
    int a = ids[r];
    if (a == 0)      { s0 += v; q0 += v * v; }
    else if (a == 1) { s1 += v; q1 += v * v; }
    else             { s2 += v; q2 += v * v; }
  }
  atomicAdd(&sum[0 * DDIM + col], s0);
  atomicAdd(&sum[1 * DDIM + col], s1);
  atomicAdd(&sum[2 * DDIM + col], s2);
  atomicAdd(&sq[0 * DDIM + col], q0);
  atomicAdd(&sq[1 * DDIM + col], q1);
  atomicAdd(&sq[2 * DDIM + col], q2);
}

__global__ void bn_apply_kernel(
    const float* __restrict__ x, const int* __restrict__ ids,
    const float* __restrict__ sum, const float* __restrict__ sq,
    const float* __restrict__ cnt, const float* __restrict__ bng,
    const float* __restrict__ bnb, float* __restrict__ outf,
    unsigned short* __restrict__ outb) {
  size_t idx = (size_t)blockIdx.x * 256 + threadIdx.x;
  if (idx >= (size_t)BDIM * DDIM) return;
  int row = (int)(idx / DDIM);
  int col = (int)(idx - (size_t)row * DDIM);
  int a = ids[row];
  float c = cnt[a];
  float safe = fmaxf(c, 1.0f);
  float mean = sum[a * DDIM + col] / safe;
  float var  = sq[a * DDIM + col] / safe - mean * mean;
  float v = x[idx];
  float nrm = (v - mean) * rsqrtf(var + LN_EPS) * bng[a * DDIM + col] +
              bnb[a * DDIM + col];
  float o = (c > 1.0f) ? nrm : v;
  outf[idx] = o;
  outb[idx] = f2bf(o);
}

// ---------------------------------------------------------------------------
// Orchestration
// ---------------------------------------------------------------------------
extern "C" void kernel_launch(void* const* d_in, const int* in_sizes, int n_in,
                              void* d_out, int out_size, void* d_ws,
                              size_t ws_size, hipStream_t stream) {
  const float* f1   = (const float*)d_in[0];
  const float* f2   = (const float*)d_in[1];
  const int*   ids  = (const int*)d_in[2];
  const float* gW1  = (const float*)d_in[3];
  const float* gb1  = (const float*)d_in[4];
  const float* gl1g = (const float*)d_in[5];
  const float* gl1b = (const float*)d_in[6];
  const float* gW2  = (const float*)d_in[7];
  const float* gb2  = (const float*)d_in[8];
  const float* gl2g = (const float*)d_in[9];
  const float* gl2b = (const float*)d_in[10];
  const float* gW3  = (const float*)d_in[11];
  const float* gb3  = (const float*)d_in[12];
  const float* qW   = (const float*)d_in[13];
  const float* qb   = (const float*)d_in[14];
  const float* kW   = (const float*)d_in[15];
  const float* kb   = (const float*)d_in[16];
  const float* vW   = (const float*)d_in[17];
  const float* vb   = (const float*)d_in[18];
  const float* alng = (const float*)d_in[19];
  const float* alnb = (const float*)d_in[20];
  const float* bng  = (const float*)d_in[21];
  const float* bnb  = (const float*)d_in[22];
  const float* ln1g = (const float*)d_in[23];
  const float* ln1b = (const float*)d_in[24];
  const float* fW1  = (const float*)d_in[25];
  const float* fb1  = (const float*)d_in[26];
  const float* fW2  = (const float*)d_in[27];
  const float* fb2  = (const float*)d_in[28];
  const float* ln2g = (const float*)d_in[29];
  const float* ln2b = (const float*)d_in[30];

  char* base = (char*)d_ws;
  size_t off = 0;
  auto takeb = [&](size_t bytes) -> char* {
    char* p = base + off;
    off += (bytes + 255) & ~(size_t)255;
    return p;
  };
  const size_t BD  = (size_t)BDIM * DDIM;     // elements
  typedef unsigned short ushort_t;

  // f32 buffers
  float* x1   = (float*)takeb(BDIM * (size_t)HDIM * 4);
  float* x2   = (float*)takeb(BDIM * (size_t)(HDIM / 2) * 4);
  float* glog = (float*)takeb(BD * 4);   // reused as attnout/gated
  float* gf   = (float*)takeb(BD * 4);
  float* q    = (float*)takeb(BD * 4);   // reused as norm
  float* k1   = (float*)takeb(BD * 4);   // reused as ffn2
  float* k2   = (float*)takeb(BD * 4);
  float* v1   = (float*)takeb(BD * 4);   // reused as res1
  float* v2   = (float*)takeb(BD * 4);
  float* ssum = (float*)takeb(3 * DDIM * 4);
  float* ssq  = (float*)takeb(3 * DDIM * 4);
  float* cnt  = (float*)takeb(256);
  // bf16 buffers
  ushort_t* catb  = (ushort_t*)takeb(BD * 2 * 2);  // reused as ffn1b
  ushort_t* x1b   = (ushort_t*)takeb(BDIM * (size_t)HDIM * 2);
  ushort_t* x2b   = (ushort_t*)takeb(BDIM * (size_t)(HDIM / 2) * 2);
  ushort_t* f1b   = (ushort_t*)takeb(BD * 2);
  ushort_t* f2b   = (ushort_t*)takeb(BD * 2);
  ushort_t* gfb   = (ushort_t*)takeb(BD * 2);
  ushort_t* normb = (ushort_t*)takeb(BD * 2);
  // bf16 transposed weights [N,K]
  ushort_t* gW1t = (ushort_t*)takeb((size_t)HDIM * 2 * DDIM * 2);
  ushort_t* gW2t = (ushort_t*)takeb((size_t)(HDIM / 2) * HDIM * 2);
  ushort_t* gW3t = (ushort_t*)takeb((size_t)DDIM * (HDIM / 2) * 2);
  ushort_t* qWt  = (ushort_t*)takeb((size_t)DDIM * DDIM * 2);
  ushort_t* kWt  = (ushort_t*)takeb((size_t)DDIM * DDIM * 2);
  ushort_t* vWt  = (ushort_t*)takeb((size_t)DDIM * DDIM * 2);
  ushort_t* fW1t = (ushort_t*)takeb((size_t)(2 * DDIM) * DDIM * 2);
  ushort_t* fW2t = (ushort_t*)takeb((size_t)DDIM * (2 * DDIM) * 2);

  float*    attnout = glog;  // glog dead after gate_combine
  float*    res1    = v1;    // v1 dead after attn_kernel
  float*    norm    = q;     // q dead after attn_kernel
  float*    ffn2    = k1;    // k1 dead after attn_kernel
  ushort_t* ffn1b   = catb;  // catb dead after gate GEMM #1
  float*    out     = (float*)d_out;

  const dim3 blk(256);
  const unsigned gBD  = (unsigned)((BD + 255) / 256);
  const unsigned gBD2 = (unsigned)((2 * BD + 255) / 256);

  // ---- one-time operand conversions (bf16 weights transposed to [N,K]) ----
  wt_transpose_kernel<<<dim3(HDIM / 32, (2 * DDIM) / 32), blk, 0, stream>>>(
      gW1, gW1t, 2 * DDIM, HDIM);
  wt_transpose_kernel<<<dim3((HDIM / 2) / 32, HDIM / 32), blk, 0, stream>>>(
      gW2, gW2t, HDIM, HDIM / 2);
  wt_transpose_kernel<<<dim3(DDIM / 32, (HDIM / 2) / 32), blk, 0, stream>>>(
      gW3, gW3t, HDIM / 2, DDIM);
  wt_transpose_kernel<<<dim3(DDIM / 32, DDIM / 32), blk, 0, stream>>>(
      qW, qWt, DDIM, DDIM);
  wt_transpose_kernel<<<dim3(DDIM / 32, DDIM / 32), blk, 0, stream>>>(
      kW, kWt, DDIM, DDIM);
  wt_transpose_kernel<<<dim3(DDIM / 32, DDIM / 32), blk, 0, stream>>>(
      vW, vWt, DDIM, DDIM);
  wt_transpose_kernel<<<dim3((2 * DDIM) / 32, DDIM / 32), blk, 0, stream>>>(
      fW1, fW1t, DDIM, 2 * DDIM);
  wt_transpose_kernel<<<dim3(DDIM / 32, (2 * DDIM) / 32), blk, 0, stream>>>(
      fW2, fW2t, 2 * DDIM, DDIM);
  cvt_bf16_kernel<<<dim3(gBD), blk, 0, stream>>>(f1, f1b, BD);
  cvt_bf16_kernel<<<dim3(gBD), blk, 0, stream>>>(f2, f2b, BD);
  concat_bf16_kernel<<<dim3(gBD2), blk, 0, stream>>>(f1, f2, catb);

  // ---- gate MLP ----
  gemm_bf16_kernel<0, 0><<<dim3(HDIM / 128, BDIM / 128), blk, 0, stream>>>(
      catb, gW1t, gb1, x1, BDIM, HDIM, 2 * DDIM);
  ln_kernel<<<dim3(BDIM), blk, 0, stream>>>(x1, nullptr, gl1g, gl1b, nullptr,
                                            x1b, HDIM, 1);
  gemm_bf16_kernel<0, 0><<<dim3((HDIM / 2) / 128, BDIM / 128), blk, 0, stream>>>(
      x1b, gW2t, gb2, x2, BDIM, HDIM / 2, HDIM);
  ln_kernel<<<dim3(BDIM), blk, 0, stream>>>(x2, nullptr, gl2g, gl2b, nullptr,
                                            x2b, HDIM / 2, 1);
  gemm_bf16_kernel<0, 0><<<dim3(DDIM / 128, BDIM / 128), blk, 0, stream>>>(
      x2b, gW3t, gb3, glog, BDIM, DDIM, HDIM / 2);
  gate_combine_kernel<<<dim3(gBD), blk, 0, stream>>>(glog, f1, f2, gf, gfb);

  // ---- q/k/v projections ----
  gemm_bf16_kernel<0, 0><<<dim3(DDIM / 128, BDIM / 128), blk, 0, stream>>>(
      gfb, qWt, qb, q, BDIM, DDIM, DDIM);
  gemm_bf16_kernel<0, 0><<<dim3(DDIM / 128, BDIM / 128), blk, 0, stream>>>(
      f1b, kWt, kb, k1, BDIM, DDIM, DDIM);
  gemm_bf16_kernel<0, 0><<<dim3(DDIM / 128, BDIM / 128), blk, 0, stream>>>(
      f2b, kWt, kb, k2, BDIM, DDIM, DDIM);
  gemm_bf16_kernel<0, 0><<<dim3(DDIM / 128, BDIM / 128), blk, 0, stream>>>(
      f1b, vWt, vb, v1, BDIM, DDIM, DDIM);
  gemm_bf16_kernel<0, 0><<<dim3(DDIM / 128, BDIM / 128), blk, 0, stream>>>(
      f2b, vWt, vb, v2, BDIM, DDIM, DDIM);

  // ---- attention combine + residual LNs ----
  attn_kernel<<<dim3(BDIM), blk, 0, stream>>>(q, k1, k2, v1, v2, gf, attnout);
  ln_kernel<<<dim3(BDIM), blk, 0, stream>>>(attnout, nullptr, alng, alnb,
                                            attnout, nullptr, DDIM, 0);
  ln_kernel<<<dim3(BDIM), blk, 0, stream>>>(attnout, f1, ln1g, ln1b, res1,
                                            nullptr, DDIM, 0);

  // ---- aspect batch norm ----
  zero_kernel<<<dim3((3 * DDIM + 255) / 256), blk, 0, stream>>>(ssum, 3 * DDIM);
  zero_kernel<<<dim3((3 * DDIM + 255) / 256), blk, 0, stream>>>(ssq, 3 * DDIM);
  bn_count_kernel<<<dim3(1), blk, 0, stream>>>(ids, cnt);
  bn_stats_kernel<<<dim3(DDIM / 256, BDIM / 256), blk, 0, stream>>>(res1, ids,
                                                                    ssum, ssq);
  bn_apply_kernel<<<dim3(gBD), blk, 0, stream>>>(res1, ids, ssum, ssq, cnt,
                                                 bng, bnb, norm, normb);

  // ---- FFN (GELU + bf16 store fused in first GEMM) + final LN ----
  gemm_bf16_kernel<1, 1><<<dim3((2 * DDIM) / 128, BDIM / 128), blk, 0, stream>>>(
      normb, fW1t, fb1, ffn1b, BDIM, 2 * DDIM, DDIM);
  gemm_bf16_kernel<0, 0><<<dim3(DDIM / 128, BDIM / 128), blk, 0, stream>>>(
      ffn1b, fW2t, fb2, ffn2, BDIM, DDIM, 2 * DDIM);
  ln_kernel<<<dim3(BDIM), blk, 0, stream>>>(ffn2, norm, ln2g, ln2b, out,
                                            nullptr, DDIM, 0);
}
